// BahdanauAttention_60215441490451
// MI455X (gfx1250) — compile-verified, matrix operand order checked by reading, not run on hardware
//
#include <hip/hip_runtime.h>
#include <hip/hip_bf16.h>

// ---------------- types for WMMA ----------------
typedef __attribute__((ext_vector_type(16))) __bf16 v16bf;
typedef __attribute__((ext_vector_type(8)))  float  v8f;

#define Bsz   32
#define Ssz   2048
#define Dsz   1024
#define Usz   1024

#if defined(__has_builtin)
#if __has_builtin(__builtin_amdgcn_sched_barrier)
#define SCHED_FENCE() __builtin_amdgcn_sched_barrier(0)
#endif
#endif
#ifndef SCHED_FENCE
#define SCHED_FENCE() asm volatile("" ::: "memory")
#endif

// round-to-nearest-even-ish fp32 -> bf16 pair packer
__device__ __forceinline__ unsigned int bf16pk(float lo, float hi) {
    unsigned int a = __float_as_uint(lo), b = __float_as_uint(hi);
    a += 0x7FFFu + ((a >> 16) & 1u);
    b += 0x7FFFu + ((b >> 16) & 1u);
    return (a >> 16) | (b & 0xFFFF0000u);
}

// ---------------------------------------------------------------------------
// Kernel A: pack W1 [D=K rows][U=N cols] fp32 -> bf16 in WMMA B-fragment order.
// Fragment block (bk,bn) covers K in [bk*32,bk*32+32), N in [bn*16,bn*16+16).
// Within a block: 32 lanes x 16 bf16, contiguous per lane (32B per lane).
// Lane L<16:  N = bn*16+L,    K = bk*32 + {0..7, 16..23}
// Lane L>=16: N = bn*16+L-16, K = bk*32 + {8..15, 24..31}
// ---------------------------------------------------------------------------
__global__ void pack_w1_kernel(const float* __restrict__ W1,
                               unsigned short* __restrict__ W1p) {
    int p    = blockIdx.x * 256 + threadIdx.x;      // 0 .. 1M-1
    int e    = p & 15;
    int lane = (p >> 4) & 31;
    int blk  = p >> 9;                               // 0 .. 2047
    int bn   = blk & 63;
    int bk   = blk >> 6;
    int n    = bn * 16 + (lane & 15);
    int kh   = lane >> 4;
    int k    = bk * 32 + ((e < 8) ? (kh * 8 + e) : (16 + kh * 8 + (e - 8)));
    unsigned int u = __float_as_uint(W1[(size_t)k * Usz + n]);
    u += 0x7FFFu + ((u >> 16) & 1u);
    W1p[p] = (unsigned short)(u >> 16);
}

// ---------------------------------------------------------------------------
// Kernel B: Qb[b,u] = query@W2 + b1 + b2 + b3  (tiny GEMV, 67 MFLOP)
// ---------------------------------------------------------------------------
__global__ void qproj_kernel(const float* __restrict__ query,
                             const float* __restrict__ W2,
                             const float* __restrict__ b1,
                             const float* __restrict__ b2,
                             const float* __restrict__ b3,
                             float* __restrict__ Qb) {
    int idx = blockIdx.x * 256 + threadIdx.x;       // 0 .. 32767
    int b = idx >> 10, u = idx & 1023;
    float acc = b1[u] + b2[u] + b3[u];
    const float* q = query + (size_t)b * Dsz;
    #pragma unroll 4
    for (int d = 0; d < Dsz; ++d)
        acc += q[d] * W2[(size_t)d * Usz + u];
    Qb[idx] = acc;
}

// ---------------------------------------------------------------------------
// Kernel C: fused  scores[b,s] = tanh(values@W1 + Qb + cov*W3) @ V  (+ bV)
// Block = 256 threads = 8 waves = 2 row-tiles(16) x 4 col-groups(256 cols).
// A tile staged once per K-step in fragment-ordered LDS (double-buffered).
// B fragments stream through a ping-pong pair of 4-fragment register groups
// with sched_barriers so ~8 global_load_b128 stay in flight behind each
// 4-WMMA burst (defeats the scheduler's just-in-time load sinking).
// ---------------------------------------------------------------------------
__global__ __launch_bounds__(256, 1)
void fused_score_kernel(const float* __restrict__ values,
                        const float* __restrict__ coverage,
                        const float* __restrict__ W3,
                        const float* __restrict__ V,
                        const float* __restrict__ bV,
                        const float* __restrict__ Qb,
                        const unsigned short* __restrict__ W1p,
                        float* __restrict__ scores) {
    // double-buffered A tile: 2 bufs x 2 rowTiles x 32 lanes x 16 bf16 = 4KB
    __shared__ __attribute__((aligned(32))) unsigned short ldsA[2048];
    __shared__ float ldsScore[32];

    const int tid = threadIdx.x;
    if (tid < 32) ldsScore[tid] = 0.0f;

    const int wg   = blockIdx.x;
    const int b    = (wg * 32) >> 11;          // 2048 rows per batch
    const int s0wg = (wg * 32) & 2047;
    const int wave = tid >> 5;
    const int lane = tid & 31;
    const int rowTile  = wave & 1;
    const int colBase  = (wave >> 1) * 256;
    const int s0   = s0wg + rowTile * 16;
    const int mB   = (lane < 16) ? 0 : 8;

    // ---- staging role: this thread owns elements e4..e4+3 of one fragment ----
    const int srt  = tid >> 7;            // staged rowTile
    const int sr   = (tid >> 2) & 31;     // staged lane
    const int se4  = (tid & 3) * 4;       // staged element quad
    const int skh  = sr >> 4;
    const int skofs = (se4 < 8) ? (skh * 8 + se4) : (16 + skh * 8 + (se4 - 8));
    const float* stageSrc =
        values + ((size_t)b * Ssz + s0wg + srt * 16 + (sr & 15)) * (size_t)Dsz + skofs;
    const int stageDst = (srt * 32 + sr) * 16 + se4;   // + buf*1024

    // prologue: stage K-block 0 into buffer 0
    {
        float4 f = *(const float4*)(stageSrc);
        uint2 p; p.x = bf16pk(f.x, f.y); p.y = bf16pk(f.z, f.w);
        *(uint2*)&ldsA[stageDst] = p;
    }
    __syncthreads();

    v8f acc[16];
    #pragma unroll
    for (int t = 0; t < 16; ++t) { v8f z = {}; acc[t] = z; }

#define LOADG(dst, base)                                                      \
    _Pragma("unroll")                                                         \
    for (int i = 0; i < 4; ++i)                                               \
        dst[i] = *(const v16bf*)(bPtr + ((base) + i) * 512);

#define WMMAG(src, base)                                                      \
    _Pragma("unroll")                                                         \
    for (int i = 0; i < 4; ++i)                                               \
        acc[(base) + i] = __builtin_amdgcn_wmma_f32_16x16x32_bf16(            \
            false, aFrag, false, src[i], (short)0, acc[(base) + i],           \
            false, false);

    int buf = 0;
    for (int k0 = 0; k0 < Dsz; k0 += 32) {
        const bool more = (k0 + 32) < Dsz;
        float4 f = {0.f, 0.f, 0.f, 0.f};
        if (more) f = *(const float4*)(stageSrc + k0 + 32);   // overlap with WMMAs

        v16bf aFrag = *(const v16bf*)&ldsA[(size_t)buf * 1024 +
                                           (rowTile * 32 + lane) * 16];
        const unsigned short* bPtr =
            W1p + ((size_t)((k0 >> 5) * 64 + (colBase >> 4))) * 512 + lane * 16;

        // ---- ping-pong B pipeline: 8 loads always in flight per WMMA burst ----
        v16bf b0[4], b1[4];
        LOADG(b0, 0)
        LOADG(b1, 4)
        SCHED_FENCE();
        WMMAG(b0, 0)
        SCHED_FENCE();
        LOADG(b0, 8)
        SCHED_FENCE();
        WMMAG(b1, 4)
        SCHED_FENCE();
        LOADG(b1, 12)
        SCHED_FENCE();
        WMMAG(b0, 8)
        SCHED_FENCE();
        WMMAG(b1, 12)
        SCHED_FENCE();

        // ---- stage next K-block into the other buffer ----
        if (more) {
            uint2 p; p.x = bf16pk(f.x, f.y); p.y = bf16pk(f.z, f.w);
            *(uint2*)&ldsA[(size_t)(buf ^ 1) * 1024 + stageDst] = p;
        }
        __syncthreads();
        buf ^= 1;
    }
#undef LOADG
#undef WMMAG

    // ---- epilogue: + Qb + cov*W3, tanh, dot with V ----
    float covRow[8];
    #pragma unroll
    for (int j = 0; j < 8; ++j)
        covRow[j] = coverage[(size_t)b * Ssz + s0 + mB + j];

    float part[8];
    #pragma unroll
    for (int j = 0; j < 8; ++j) part[j] = 0.0f;

    #pragma unroll
    for (int t = 0; t < 16; ++t) {
        int n = colBase + t * 16 + (lane & 15);
        float qn  = Qb[b * Usz + n];
        float w3n = W3[n];
        float vn  = V[n];
        #pragma unroll
        for (int j = 0; j < 8; ++j) {
            float val = acc[t][j] + qn + covRow[j] * w3n;
            part[j] += tanhf(val) * vn;
        }
    }

    // reduce over the 16-lane N group (butterfly keeps halves separate)
    #pragma unroll
    for (int j = 0; j < 8; ++j) {
        float p = part[j];
        p += __shfl_xor(p, 1, 32);
        p += __shfl_xor(p, 2, 32);
        p += __shfl_xor(p, 4, 32);
        p += __shfl_xor(p, 8, 32);
        part[j] = p;
    }
    if ((lane & 15) == 0) {
        #pragma unroll
        for (int j = 0; j < 8; ++j)
            atomicAdd(&ldsScore[rowTile * 16 + mB + j], part[j]);  // ds_add_f32
    }
    __syncthreads();
    if (tid < 32)
        scores[(size_t)b * Ssz + s0wg + tid] = ldsScore[tid] + bV[0];
}

// ---------------------------------------------------------------------------
// Kernel D: softmax over S per batch; emit attention weights + new coverage
// ---------------------------------------------------------------------------
__global__ void softmax_kernel(const float* __restrict__ scores,
                               const float* __restrict__ coverage,
                               float* __restrict__ att,
                               float* __restrict__ ncov,
                               float* __restrict__ wgt) {
    __shared__ float red[256];
    const int b = blockIdx.x, tid = threadIdx.x;
    const float* sc = scores + (size_t)b * Ssz;

    float lmax = -3.4e38f;
    for (int i = tid; i < Ssz; i += 256) lmax = fmaxf(lmax, sc[i]);
    red[tid] = lmax; __syncthreads();
    for (int o = 128; o > 0; o >>= 1) {
        if (tid < o) red[tid] = fmaxf(red[tid], red[tid + o]);
        __syncthreads();
    }
    const float mx = red[0];
    __syncthreads();

    float ls = 0.0f;
    for (int i = tid; i < Ssz; i += 256) ls += __expf(sc[i] - mx);
    red[tid] = ls; __syncthreads();
    for (int o = 128; o > 0; o >>= 1) {
        if (tid < o) red[tid] += red[tid + o];
        __syncthreads();
    }
    const float inv = 1.0f / red[0];

    for (int i = tid; i < Ssz; i += 256) {
        float w = __expf(sc[i] - mx) * inv;
        size_t idx = (size_t)b * Ssz + i;
        att[idx]  = w;
        wgt[idx]  = w;
        ncov[idx] = coverage[idx] + w;
    }
}

// ---------------------------------------------------------------------------
// Kernel E: context[b,d] = sum_s w[b,s] * values[b,s,d]  (coalesced over d)
// ---------------------------------------------------------------------------
__global__ __launch_bounds__(256)
void context_kernel(const float* __restrict__ values,
                    const float* __restrict__ wgt,
                    float* __restrict__ ctx) {
    int b = blockIdx.x >> 2;
    int d = ((blockIdx.x & 3) << 8) + threadIdx.x;
    const float* vb = values + (size_t)b * Ssz * Dsz + d;
    const float* wb = wgt + (size_t)b * Ssz;
    float a0 = 0.f, a1 = 0.f, a2 = 0.f, a3 = 0.f;
    for (int s = 0; s < Ssz; s += 4) {
        a0 += wb[s + 0] * vb[(size_t)(s + 0) * Dsz];
        a1 += wb[s + 1] * vb[(size_t)(s + 1) * Dsz];
        a2 += wb[s + 2] * vb[(size_t)(s + 2) * Dsz];
        a3 += wb[s + 3] * vb[(size_t)(s + 3) * Dsz];
    }
    ctx[(size_t)b * Dsz + d] = (a0 + a1) + (a2 + a3);
}

// ---------------------------------------------------------------------------
extern "C" void kernel_launch(void* const* d_in, const int* in_sizes, int n_in,
                              void* d_out, int out_size, void* d_ws, size_t ws_size,
                              hipStream_t stream) {
    const float* query    = (const float*)d_in[0];
    const float* values   = (const float*)d_in[1];
    const float* coverage = (const float*)d_in[2];
    const float* W1       = (const float*)d_in[3];
    const float* b1       = (const float*)d_in[4];
    const float* W2       = (const float*)d_in[5];
    const float* b2       = (const float*)d_in[6];
    const float* W3       = (const float*)d_in[7];
    const float* b3       = (const float*)d_in[8];
    const float* V        = (const float*)d_in[9];
    const float* bV       = (const float*)d_in[10];

    // workspace layout (floats): Qb[32768] | scores[65536] | wgt[65536] | W1p(bf16 1M)
    float* ws      = (float*)d_ws;
    float* Qb      = ws;
    float* scores  = ws + 32768;
    float* wgt     = ws + 32768 + 65536;
    unsigned short* W1p = (unsigned short*)(ws + 32768 + 65536 + 65536);

    // outputs: context[32768] | attention[65536] | new_coverage[65536]
    float* ctx  = (float*)d_out;
    float* att  = ctx + Bsz * Dsz;
    float* ncov = att + Bsz * Ssz;

    pack_w1_kernel    <<<4096, 256, 0, stream>>>(W1, W1p);
    qproj_kernel      <<< 128, 256, 0, stream>>>(query, W2, b1, b2, b3, Qb);
    fused_score_kernel<<<2048, 256, 0, stream>>>(values, coverage, W3, V, bV,
                                                 Qb, W1p, scores);
    softmax_kernel    <<<  Bsz, 256, 0, stream>>>(scores, coverage, att, ncov, wgt);
    context_kernel    <<< 128, 256, 0, stream>>>(values, wgt, ctx);
}
